// Physical3DBranch_9131100472089
// MI455X (gfx1250) — matrix operand order repaired and checked by reading, not compiled, not run on hardware
//
#include <hip/hip_runtime.h>
#include <hip/hip_bf16.h>

#define HIDN    256
#define NLAY    4
#define EMB     16
#define EDIM    16
#define NNODE   10000
#define NPAD    10016         // NNODE padded to multiple of 32 for M-blocked GEMM
#define NEDGE   160000
#define EINW    544           // 2*HID + EMB + EDIM
#define CATW    512           // 2*HID
#define INVKP   288           // HID+4 = 260 padded to mult of 32
#define CUT     10.0f
#define PI_F    3.14159265358979323846f

typedef __attribute__((ext_vector_type(16))) __bf16 v16bf;
typedef __attribute__((ext_vector_type(8)))  float  v8f;
typedef __attribute__((ext_vector_type(4)))  unsigned int v4u;

union Frag {
    struct { v4u lo, hi; } q;   // 2 x 16B chunks
    v16bf v;                    // 16 bf16 = 8 VGPRs
};

__device__ __forceinline__ float siluf(float x) { return x / (1.0f + __expf(-x)); }

// ---------------------------------------------------------------------------
// Register-blocked WMMA GEMM: each wave computes a 32x64 block of C
// (2 M-tiles x 4 N-tiles -> 8 wmma per K-step, 12 b128 loads per K-step).
// A[M,Kpad] bf16 row-major; B transposed: BT[Nn,Kpad] bf16.
// ---------------------------------------------------------------------------
enum { M_SILU_BF16 = 0, M_SILU_W_BF16 = 1, M_SILU_F32 = 2, M_PLAIN_F32 = 3, M_RES_F32 = 4 };

template<int MODE>
__global__ __launch_bounds__(256)
void k_gemm(const __bf16* __restrict__ A, int Kpad,
            const __bf16* __restrict__ BT,
            const float* __restrict__ bias,
            void* __restrict__ outp,
            const float* __restrict__ wrow,     // MODE==1: per-row scale
            const float* __restrict__ resid,    // MODE==4: residual
            int M, int Nn, int totalWaves) {
    const int wave = blockIdx.x * 8 + (threadIdx.x >> 5);
    if (wave >= totalWaves) return;
    const int lane = threadIdx.x & 31;
    const int nb   = Nn >> 6;                 // number of 64-wide col blocks
    const int mblk = wave / nb;
    const int nblk = wave - mblk * nb;
    const int grp  = lane >> 4;               // lane half
    const int ln   = lane & 15;

    // A 16x32 bf16 frag: lanes 0-15 hold K {0..7,16..23}, lanes 16-31 {8..15,24..31}
    const __bf16* a0 = A + (size_t)(mblk * 32 + ln) * Kpad + grp * 8;
    const __bf16* a1 = a0 + (size_t)16 * Kpad;
    // B 32x16 bf16 frag: lanes 0-15 hold K {0..7,16..23}, lanes 16-31 {8..15,24..31}
    const __bf16* b0 = BT + (size_t)(nblk * 64 + ln) * Kpad + grp * 8;
    const size_t  bstride = (size_t)16 * Kpad;

    v8f acc[2][4];
#pragma unroll
    for (int mi = 0; mi < 2; ++mi)
#pragma unroll
        for (int j = 0; j < 4; ++j)
            acc[mi][j] = (v8f){0.f, 0.f, 0.f, 0.f, 0.f, 0.f, 0.f, 0.f};

    for (int k = 0; k < Kpad; k += 32) {
        Frag fa0, fa1, fb[4];
        fa0.q.lo = *reinterpret_cast<const v4u*>(a0 + k);
        fa0.q.hi = *reinterpret_cast<const v4u*>(a0 + k + 16);
        fa1.q.lo = *reinterpret_cast<const v4u*>(a1 + k);
        fa1.q.hi = *reinterpret_cast<const v4u*>(a1 + k + 16);
#pragma unroll
        for (int j = 0; j < 4; ++j) {
            fb[j].q.lo = *reinterpret_cast<const v4u*>(b0 + j * bstride + k);
            fb[j].q.hi = *reinterpret_cast<const v4u*>(b0 + j * bstride + k + 16);
        }
#pragma unroll
        for (int j = 0; j < 4; ++j) {
            acc[0][j] = __builtin_amdgcn_wmma_f32_16x16x32_bf16(false, fa0.v, false, fb[j].v,
                                                               (short)0, acc[0][j], false, false);
            acc[1][j] = __builtin_amdgcn_wmma_f32_16x16x32_bf16(false, fa1.v, false, fb[j].v,
                                                               (short)0, acc[1][j], false, false);
        }
    }

#pragma unroll
    for (int mi = 0; mi < 2; ++mi) {
#pragma unroll
        for (int j = 0; j < 4; ++j) {
            const int col = nblk * 64 + j * 16 + ln;
            const float b = bias[col];
#pragma unroll
            for (int i = 0; i < 8; ++i) {
                const int r = mblk * 32 + mi * 16 + grp * 8 + i;  // C/D: vgpr i -> row i (+8 hi lanes)
                if (r < M) {
                    const size_t o = (size_t)r * Nn + col;
                    float x = acc[mi][j][i] + b;
                    if (MODE == M_SILU_BF16)        ((__bf16*)outp)[o] = (__bf16)siluf(x);
                    else if (MODE == M_SILU_W_BF16) ((__bf16*)outp)[o] = (__bf16)(siluf(x) * wrow[r]);
                    else if (MODE == M_SILU_F32)    ((float*)outp)[o]  = siluf(x);
                    else if (MODE == M_PLAIN_F32)   ((float*)outp)[o]  = x;
                    else                            ((float*)outp)[o]  = resid[o] + x;
                }
            }
        }
    }
}

// ---------------------------------------------------------------------------
// weight transpose + fp32->bf16: src[K,N] -> dst[N,Kpad] (zero pad K..Kpad)
// ---------------------------------------------------------------------------
__global__ void k_wT(const float* __restrict__ src, __bf16* __restrict__ dst,
                     int K, int N, int Kpad) {
    int idx = blockIdx.x * 256 + threadIdx.x;
    if (idx >= N * Kpad) return;
    int n = idx / Kpad, k = idx - n * Kpad;
    dst[idx] = (k < K) ? (__bf16)src[(size_t)k * N + n] : (__bf16)0.f;
}

// ---------------------------------------------------------------------------
// per-edge gather: build A_edge[E,544] bf16 = [h[row] | h[col] | sin-emb | edge_attr]
// also rel[E,3], dist[E], cutoff weight w[E].  One block per edge, 256 threads.
// ---------------------------------------------------------------------------
__global__ __launch_bounds__(256)
void k_edge_build(const float* __restrict__ h, const float* __restrict__ pos,
                  const int* __restrict__ row, const int* __restrict__ col,
                  const float* __restrict__ eattr,
                  __bf16* __restrict__ Ae, float* __restrict__ rel,
                  float* __restrict__ dist, float* __restrict__ wv) {
    const int e = blockIdx.x;
    const int t = threadIdx.x;
    const int r = row[e], c = col[e];
    const float rx = pos[r*3+0] - pos[c*3+0];
    const float ry = pos[r*3+1] - pos[c*3+1];
    const float rz = pos[r*3+2] - pos[c*3+2];
    const float d  = sqrtf(rx*rx + ry*ry + rz*rz + 1e-8f);
    __bf16* arow = Ae + (size_t)e * EINW;
    arow[t]         = (__bf16)h[(size_t)r * HIDN + t];
    arow[HIDN + t]  = (__bf16)h[(size_t)c * HIDN + t];
    if (t < EMB) {
        const float fr = PI_F * (float)(t + 1) / CUT;
        arow[2*HIDN + t] = (__bf16)__sinf(d * fr);
    } else if (t < 2*EMB) {
        arow[2*HIDN + t] = (__bf16)eattr[(size_t)e * EDIM + (t - EMB)];
    }
    if (t == 0) {
        rel[e*3+0] = rx; rel[e*3+1] = ry; rel[e*3+2] = rz;
        dist[e] = d;
        wv[e]   = 0.5f * (__cosf(PI_F * fminf(d, CUT) / CUT) + 1.0f);
    }
}

// cx[e] = tanh( x1[e,:] . Wx2 )  -- one wave per edge
__global__ __launch_bounds__(256)
void k_cx(const float* __restrict__ x1, const float* __restrict__ wx2,
          float* __restrict__ cx, int E) {
    const int w = blockIdx.x * 8 + (threadIdx.x >> 5);
    if (w >= E) return;
    const int lane = threadIdx.x & 31;
    const float* xr = x1 + (size_t)w * HIDN;
    float s = 0.f;
#pragma unroll
    for (int j = lane; j < HIDN; j += 32) s += xr[j] * wx2[j];
#pragma unroll
    for (int off = 16; off > 0; off >>= 1) s += __shfl_down(s, off, 32);
    if (lane == 0) cx[w] = tanhf(s);
}

__global__ void k_pos_scatter(const float* __restrict__ rel, const float* __restrict__ dist,
                              const float* __restrict__ cx, const int* __restrict__ row,
                              float* __restrict__ posn, int E) {
    int e = blockIdx.x * 256 + threadIdx.x;
    if (e >= E) return;
    float cscale = cx[e] / (dist[e] + 1.0f);
    int r = row[e];
    atomicAdd(&posn[r*3+0], rel[e*3+0] * cscale);
    atomicAdd(&posn[r*3+1], rel[e*3+1] * cscale);
    atomicAdd(&posn[r*3+2], rel[e*3+2] * cscale);
}

__global__ __launch_bounds__(256)
void k_agg_scatter(const __bf16* __restrict__ m, const int* __restrict__ row,
                   float* __restrict__ agg) {
    int e = blockIdx.x, t = threadIdx.x;
    int r = row[e];
    atomicAdd(&agg[(size_t)r * HIDN + t], (float)m[(size_t)e * HIDN + t]);
}

__global__ void k_cat(const float* __restrict__ h, const float* __restrict__ agg,
                      __bf16* __restrict__ cat) {
    int i = blockIdx.x, t = threadIdx.x;
    cat[(size_t)i*CATW + t]        = (__bf16)h[(size_t)i*HIDN + t];
    cat[(size_t)i*CATW + HIDN + t] = (__bf16)agg[(size_t)i*HIDN + t];
}

__global__ void k_f2bf(const float* src, __bf16* dst, int n) {
    int i = blockIdx.x * 256 + threadIdx.x;
    if (i < n) dst[i] = (__bf16)src[i];
}
__global__ void k_zero(float* p, int n) {
    int i = blockIdx.x * 256 + threadIdx.x;
    if (i < n) p[i] = 0.f;
}
__global__ void k_copy(float* d, const float* s, int n) {
    int i = blockIdx.x * 256 + threadIdx.x;
    if (i < n) d[i] = s[i];
}

__global__ __launch_bounds__(256)
void k_pos_mean(const float* __restrict__ pos, float* __restrict__ mean) {
    __shared__ float sm[3][256];
    int t = threadIdx.x;
    float sx = 0.f, sy = 0.f, sz = 0.f;
    for (int i = t; i < NNODE; i += 256) {
        sx += pos[i*3+0]; sy += pos[i*3+1]; sz += pos[i*3+2];
    }
    sm[0][t] = sx; sm[1][t] = sy; sm[2][t] = sz;
    __syncthreads();
    for (int s2 = 128; s2 > 0; s2 >>= 1) {
        if (t < s2) { sm[0][t] += sm[0][t+s2]; sm[1][t] += sm[1][t+s2]; sm[2][t] += sm[2][t+s2]; }
        __syncthreads();
    }
    if (t == 0) {
        mean[0] = sm[0][0] / (float)NNODE;
        mean[1] = sm[1][0] / (float)NNODE;
        mean[2] = sm[2][0] / (float)NNODE;
    }
}

// inv_in[N,288] = [h(256) | ||pos|| | mean(3) | zero-pad(28)] bf16, one block/node, 288 threads
__global__ void k_inv_build(const float* __restrict__ h, const float* __restrict__ pos,
                            const float* __restrict__ mean, __bf16* __restrict__ inv) {
    int i = blockIdx.x, t = threadIdx.x;
    __bf16 v;
    if (t < HIDN) v = (__bf16)h[(size_t)i * HIDN + t];
    else if (t == HIDN) {
        float px = pos[i*3], py = pos[i*3+1], pz = pos[i*3+2];
        v = (__bf16)sqrtf(px*px + py*py + pz*pz);
    } else if (t <= HIDN + 3) v = (__bf16)mean[t - HIDN - 1];
    else v = (__bf16)0.f;
    inv[(size_t)i * INVKP + t] = v;
}

// ---------------------------------------------------------------------------
extern "C" void kernel_launch(void* const* d_in, const int* in_sizes, int n_in,
                              void* d_out, int out_size, void* d_ws, size_t ws_size,
                              hipStream_t stream) {
    (void)in_sizes; (void)n_in; (void)out_size; (void)ws_size;
    // setup_inputs dict order, params flattened in insertion order
    const float* h_in   = (const float*)d_in[0];
    const float* pos_in = (const float*)d_in[1];
    const float* We1 = (const float*)d_in[2];   const float* be1 = (const float*)d_in[3];
    const float* We2 = (const float*)d_in[4];   const float* be2 = (const float*)d_in[5];
    const float* Wx1 = (const float*)d_in[6];   const float* bx1 = (const float*)d_in[7];
    const float* Wx2 = (const float*)d_in[8];
    const float* Wh1 = (const float*)d_in[9];   const float* bh1 = (const float*)d_in[10];
    const float* Wh2 = (const float*)d_in[11];  const float* bh2 = (const float*)d_in[12];
    const float* Wg1 = (const float*)d_in[13];  const float* bg1 = (const float*)d_in[14];
    const float* Wg2 = (const float*)d_in[15];  const float* bg2 = (const float*)d_in[16];
    const float* Wi1 = (const float*)d_in[17];  const float* bi1 = (const float*)d_in[18];
    const float* Wi2 = (const float*)d_in[19];  const float* bi2 = (const float*)d_in[20];
    const int*   eidx  = (const int*)d_in[22];
    const float* eattr = (const float*)d_in[23];
    const int* row = eidx;
    const int* col = eidx + NEDGE;

    // ---- workspace carve-out (~400 MB); node-side GEMM operands padded to NPAD rows ----
    char* ws = (char*)d_ws;
    size_t off = 0;
    auto alloc = [&](size_t bytes) -> char* {
        char* p = ws + off;
        off = (off + bytes + 255) & ~(size_t)255;
        return p;
    };
    char*   Abig  = alloc((size_t)NEDGE * EINW * 2);        // A_edge bf16; later x1 fp32 (smaller)
    __bf16* m1    = (__bf16*)alloc((size_t)NEDGE * HIDN * 2);
    __bf16* mbf   = (__bf16*)alloc((size_t)NEDGE * HIDN * 2);
    float*  relb  = (float*)alloc((size_t)NEDGE * 3 * 4);
    float*  distb = (float*)alloc((size_t)NEDGE * 4);
    float*  wvb   = (float*)alloc((size_t)NEDGE * 4);
    float*  cxb   = (float*)alloc((size_t)NEDGE * 4);
    float*  aggb  = (float*)alloc((size_t)NNODE * HIDN * 4);
    __bf16* catb  = (__bf16*)alloc((size_t)NPAD * CATW * 2);
    __bf16* ub    = (__bf16*)alloc((size_t)NPAD * HIDN * 2);
    float*  hb    = (float*)alloc((size_t)NPAD * HIDN * 4);
    float*  posA  = (float*)alloc((size_t)NNODE * 3 * 4);
    float*  posB  = (float*)alloc((size_t)NNODE * 3 * 4);
    __bf16* hbf   = (__bf16*)alloc((size_t)NPAD * HIDN * 2);
    __bf16* invin = (__bf16*)alloc((size_t)NPAD * INVKP * 2);
    __bf16* g1b   = (__bf16*)alloc((size_t)NPAD * 128 * 2);
    __bf16* i1b   = (__bf16*)alloc((size_t)NPAD * HIDN * 2);
    float*  meanp = (float*)alloc(256);
    __bf16* we1t  = (__bf16*)alloc((size_t)NLAY * HIDN * EINW * 2);
    __bf16* we2t  = (__bf16*)alloc((size_t)NLAY * HIDN * HIDN * 2);
    __bf16* wx1t  = (__bf16*)alloc((size_t)NLAY * HIDN * HIDN * 2);
    __bf16* wh1t  = (__bf16*)alloc((size_t)NLAY * HIDN * CATW * 2);
    __bf16* wh2t  = (__bf16*)alloc((size_t)NLAY * HIDN * HIDN * 2);
    __bf16* wg1t  = (__bf16*)alloc((size_t)128 * HIDN * 2);
    __bf16* wg2t  = (__bf16*)alloc((size_t)64 * 128 * 2);
    __bf16* wi1t  = (__bf16*)alloc((size_t)HIDN * INVKP * 2);
    __bf16* wi2t  = (__bf16*)alloc((size_t)HIDN * HIDN * 2);

    auto wconv = [&](const float* src, __bf16* dst, int K, int N2, int Kpad) {
        int tot = N2 * Kpad;
        k_wT<<<(tot + 255) / 256, 256, 0, stream>>>(src, dst, K, N2, Kpad);
    };
    for (int l = 0; l < NLAY; ++l) {
        wconv(We1 + (size_t)l*EINW*HIDN, we1t + (size_t)l*HIDN*EINW, EINW, HIDN, EINW);
        wconv(We2 + (size_t)l*HIDN*HIDN, we2t + (size_t)l*HIDN*HIDN, HIDN, HIDN, HIDN);
        wconv(Wx1 + (size_t)l*HIDN*HIDN, wx1t + (size_t)l*HIDN*HIDN, HIDN, HIDN, HIDN);
        wconv(Wh1 + (size_t)l*CATW*HIDN, wh1t + (size_t)l*HIDN*CATW, CATW, HIDN, CATW);
        wconv(Wh2 + (size_t)l*HIDN*HIDN, wh2t + (size_t)l*HIDN*HIDN, HIDN, HIDN, HIDN);
    }
    wconv(Wg1, wg1t, HIDN, 128, HIDN);
    wconv(Wg2, wg2t, 128, 64, 128);
    wconv(Wi1, wi1t, HIDN + 4, HIDN, INVKP);
    wconv(Wi2, wi2t, HIDN, HIDN, HIDN);

    // init running state
    k_copy<<<(NNODE*HIDN + 255)/256, 256, 0, stream>>>(hb, h_in, NNODE*HIDN);
    k_copy<<<(NNODE*3 + 255)/256, 256, 0, stream>>>(posA, pos_in, NNODE*3);

#define GEMM(MODE, Ap, Kp, Bp, biasp, outp, wp, rp, Mm, NnV)                                   \
    do {                                                                                       \
        int waves_ = (((Mm) + 31) / 32) * ((NnV) / 64);                                        \
        k_gemm<MODE><<<(waves_ + 7) / 8, 256, 0, stream>>>((const __bf16*)(Ap), (Kp),          \
            (const __bf16*)(Bp), (biasp), (void*)(outp), (wp), (rp), (Mm), (NnV), waves_);     \
    } while (0)

    for (int l = 0; l < NLAY; ++l) {
        __bf16* Ae = (__bf16*)Abig;
        float*  x1 = (float*)Abig;     // reuse after GEMM1 consumes Ae
        // edge features
        k_edge_build<<<NEDGE, 256, 0, stream>>>(hb, posA, row, col, eattr,
                                                Ae, relb, distb, wvb);
        // m1 = silu(Ae @ We1 + be1)
        GEMM(M_SILU_BF16, Ae, EINW, we1t + (size_t)l*HIDN*EINW, be1 + (size_t)l*HIDN,
             m1, nullptr, nullptr, NEDGE, HIDN);
        // m = silu(m1 @ We2 + be2) * w
        GEMM(M_SILU_W_BF16, m1, HIDN, we2t + (size_t)l*HIDN*HIDN, be2 + (size_t)l*HIDN,
             mbf, wvb, nullptr, NEDGE, HIDN);
        // x1 = silu(m @ Wx1 + bx1)  (fp32, aliases Abig)
        GEMM(M_SILU_F32, mbf, HIDN, wx1t + (size_t)l*HIDN*HIDN, bx1 + (size_t)l*HIDN,
             x1, nullptr, nullptr, NEDGE, HIDN);
        // cx = tanh(x1 . Wx2)
        k_cx<<<(NEDGE + 7)/8, 256, 0, stream>>>(x1, Wx2 + (size_t)l*HIDN, cxb, NEDGE);
        // pos update (double-buffered scatter)
        k_copy<<<(NNODE*3 + 255)/256, 256, 0, stream>>>(posB, posA, NNODE*3);
        k_pos_scatter<<<(NEDGE + 255)/256, 256, 0, stream>>>(relb, distb, cxb, row, posB, NEDGE);
        { float* t = posA; posA = posB; posB = t; }
        // agg = segment_sum(m, row)
        k_zero<<<(NNODE*HIDN + 255)/256, 256, 0, stream>>>(aggb, NNODE*HIDN);
        k_agg_scatter<<<NEDGE, 256, 0, stream>>>(mbf, row, aggb);
        // node update: h += silu([h|agg] @ Wh1 + bh1) @ Wh2 + bh2
        k_cat<<<NNODE, 256, 0, stream>>>(hb, aggb, catb);
        GEMM(M_SILU_BF16, catb, CATW, wh1t + (size_t)l*HIDN*CATW, bh1 + (size_t)l*HIDN,
             ub, nullptr, nullptr, NNODE, HIDN);
        GEMM(M_RES_F32, ub, HIDN, wh2t + (size_t)l*HIDN*HIDN, bh2 + (size_t)l*HIDN,
             hb, nullptr, hb, NNODE, HIDN);
    }

    // ---- outputs ----
    float* outH   = (float*)d_out;
    float* outPos = outH + (size_t)NNODE * HIDN;
    float* outGeo = outPos + (size_t)NNODE * 3;
    float* outInv = outGeo + (size_t)NNODE * 64;

    k_copy<<<(NNODE*HIDN + 255)/256, 256, 0, stream>>>(outH, hb, NNODE*HIDN);
    k_copy<<<(NNODE*3 + 255)/256, 256, 0, stream>>>(outPos, posA, NNODE*3);

    // geometric head: silu(h@Wg1+bg1)@Wg2+bg2
    k_f2bf<<<(NNODE*HIDN + 255)/256, 256, 0, stream>>>(hb, hbf, NNODE*HIDN);
    GEMM(M_SILU_BF16, hbf, HIDN, wg1t, bg1, g1b, nullptr, nullptr, NNODE, 128);
    GEMM(M_PLAIN_F32, g1b, 128, wg2t, bg2, outGeo, nullptr, nullptr, NNODE, 64);

    // invariant head
    k_pos_mean<<<1, 256, 0, stream>>>(posA, meanp);
    k_inv_build<<<NNODE, 288, 0, stream>>>(hb, posA, meanp, invin);
    GEMM(M_SILU_BF16, invin, INVKP, wi1t, bi1, i1b, nullptr, nullptr, NNODE, HIDN);
    GEMM(M_PLAIN_F32, i1b, HIDN, wi2t, bi2, outInv, nullptr, nullptr, NNODE, HIDN);
#undef GEMM
}
